// MHA_KNN_V_15960098472026
// MI455X (gfx1250) — matrix operand order repaired and checked by reading, not compiled, not run on hardware
//
#include <hip/hip_runtime.h>
#include <hip/hip_bf16.h>

// ---------------------------------------------------------------------------
// Problem constants (from reference): B=4, V=4096, E=256, H=8, K=8, HD=32
// ---------------------------------------------------------------------------
#define CB 4
#define CV 4096
#define CE 256
#define CH 8
#define CK 8
#define CHD 32

typedef __attribute__((ext_vector_type(16))) __bf16 v16bf;
typedef __attribute__((ext_vector_type(8)))  __bf16 v8bf;
typedef __attribute__((ext_vector_type(8)))  float  v8f;

// Exact parameter type of the async-LDS builtin per hipcc diagnostic:
//   "__attribute__((__vector_size__(4 * sizeof(int)))) int __device__ *"
typedef int v4i __attribute__((vector_size(4 * sizeof(int))));
#define GAS __attribute__((address_space(1)))   // global  (printed as __device__)
#define LAS __attribute__((address_space(3)))   // LDS     (printed as __shared__)

// CDNA5 async global->LDS path (guarded: falls back to plain copy if the
// builtin is absent on this toolchain)
#if defined(__gfx1250__) && __has_builtin(__builtin_amdgcn_global_load_async_to_lds_b128)
#define HAVE_ASYNC_LDS 1
#else
#define HAVE_ASYNC_LDS 0
#endif

__device__ __forceinline__ void async_wait0() {
#if HAVE_ASYNC_LDS
#if __has_builtin(__builtin_amdgcn_s_wait_asynccnt)
  __builtin_amdgcn_s_wait_asynccnt(0);
#else
  asm volatile("s_wait_asynccnt 0x0" ::: "memory");
#endif
#endif
}

// ---- bf16 helpers (bit-exact, no reliance on cast support) -----------------
__device__ __forceinline__ __bf16 f2bf(float f) {
  union { float f; unsigned u; } x; x.f = f;
  unsigned r = x.u + 0x7FFFu + ((x.u >> 16) & 1u);   // round to nearest even
  union { unsigned short s; __bf16 b; } y;
  y.s = (unsigned short)(r >> 16);
  return y.b;
}
__device__ __forceinline__ float bf2f(__bf16 b) {
  union { unsigned short s; __bf16 b; } x; x.b = b;
  union { unsigned u; float f; } y;
  y.u = ((unsigned)x.s) << 16;
  return y.f;
}

// ---------------------------------------------------------------------------
// Kernel 1: brute-force KNN (K=8 nearest incl. self, ascending distance)
// One thread per query point; x_v slice for the batch staged in LDS (48KB).
// ---------------------------------------------------------------------------
__global__ __launch_bounds__(256) void knn_kernel(const float* __restrict__ xv,
                                                  int* __restrict__ idx) {
  __shared__ float sx[CV * 3];
  const int blocksPerBatch = CV / 256;
  const int b = blockIdx.x / blocksPerBatch;
  const int chunk = blockIdx.x % blocksPerBatch;
  const float* xb = xv + (size_t)b * CV * 3;
  for (int i = threadIdx.x; i < CV * 3; i += 256) sx[i] = xb[i];
  __syncthreads();

  const int v = chunk * 256 + threadIdx.x;
  const float qx = sx[v * 3 + 0], qy = sx[v * 3 + 1], qz = sx[v * 3 + 2];

  float bd[CK]; int bi[CK];
#pragma unroll
  for (int i = 0; i < CK; ++i) { bd[i] = 3.4e38f; bi[i] = 0; }

  for (int j = 0; j < CV; ++j) {
    const float dx = sx[j * 3 + 0] - qx;
    const float dy = sx[j * 3 + 1] - qy;
    const float dz = sx[j * 3 + 2] - qz;
    const float d = dx * dx + dy * dy + dz * dz;
    if (d < bd[CK - 1]) {
      int pos = CK - 1;
      while (pos > 0 && bd[pos - 1] > d) {
        bd[pos] = bd[pos - 1]; bi[pos] = bi[pos - 1]; --pos;
      }
      bd[pos] = d; bi[pos] = j;
    }
  }
  int* o = idx + ((size_t)b * CV + v) * CK;
#pragma unroll
  for (int i = 0; i < CK; ++i) o[i] = bi[i];
}

// ---------------------------------------------------------------------------
// Kernel 2: fp32 -> bf16 conversion
// ---------------------------------------------------------------------------
__global__ __launch_bounds__(256) void cvt_bf16_kernel(const float* __restrict__ in,
                                                       __bf16* __restrict__ out, int n) {
  int i = blockIdx.x * 256 + threadIdx.x;
  if (i < n) out[i] = f2bf(in[i]);
}

// ---------------------------------------------------------------------------
// Kernel 3: WMMA GEMM  C[M,N] = A[M,K] * W[N,K]^T  (+ optional fp32 residual)
//
// Block = 8 waves. Wave w owns N-tile n0 = blockIdx.x*128 + w*16 and preloads
// its full B strip (16 rows x K=256 bf16 = 64 VGPRs) into registers ONCE.
// The block then loops over MT_PER_BLOCK M-tiles; the 16x256 bf16 A tile is
// double-buffered in LDS, staged with GLOBAL_LOAD_ASYNC_TO_LDS_B128
// (ASYNCcnt) and consumed via ds_load_b128 by all 8 waves.
// Per M-tile per wave: 8 x v_wmma_f32_16x16x32_bf16 vs 16 LDS reads.
// ---------------------------------------------------------------------------
#define MT_PER_BLOCK 16   // 256 M rows per block

__global__ __launch_bounds__(256) void wmma_gemm_bf16(const __bf16* __restrict__ A,
                                                      const __bf16* __restrict__ W,
                                                      float* __restrict__ C,
                                                      const float* __restrict__ resid,
                                                      int M, int N, int Kd) {
  __shared__ __align__(16) __bf16 sA[2][16 * CE];   // 2 x 8KB double buffer

  const int mBase = blockIdx.y * (MT_PER_BLOCK * 16);
  if (blockIdx.x * 128 >= N || mBase >= M) return;  // block-uniform guard

  const int wave = threadIdx.x >> 5;
  const int lane = threadIdx.x & 31;
  const int half = lane >> 4;            // 0: lanes 0-15, 1: lanes 16-31
  const int l16  = lane & 15;
  const int n0   = blockIdx.x * 128 + wave * 16;
  const int nCol = n0 + l16;

  union BF { v16bf v; v8bf h[2]; };

  // ---- preload B strip: this wave's 16 weight rows, all of K, in registers --
  BF bfrag[8];
  {
    const size_t wBase = (size_t)nCol * Kd;
#pragma unroll
    for (int ks = 0; ks < 8; ++ks) {
      // B k order used here (contiguous, same as A): lane half 0 -> K 0..15; half 1 -> K 16..31
      const int kb = ks * 32 + half * 16;
      bfrag[ks].h[0] = *(const v8bf*)(W + wBase + kb);
      bfrag[ks].h[1] = *(const v8bf*)(W + wBase + kb + 8);
    }
  }

  // cooperative A staging: 256 threads x 32 contiguous bytes (2 x b128)
  const int stR = threadIdx.x >> 4;           // 0..15 (tile row)
  const int stO = (threadIdx.x & 15) * 16;    // 0..240 (bf16 column, step 16)

  auto stage_tile = [&](int buf, int m0) {
    const __bf16* src = A + (size_t)(m0 + stR) * Kd + stO;
    __bf16* dst = &sA[buf][stR * CE + stO];
#if HAVE_ASYNC_LDS
    __builtin_amdgcn_global_load_async_to_lds_b128(
        (GAS v4i*)(src), (LAS v4i*)(dst), 0, 0);
    __builtin_amdgcn_global_load_async_to_lds_b128(
        (GAS v4i*)(src + 8), (LAS v4i*)(dst + 8), 0, 0);
#else
    *(v8bf*)dst       = *(const v8bf*)src;
    *(v8bf*)(dst + 8) = *(const v8bf*)(src + 8);
#endif
    // keep the tile after next warming in L2 (global_prefetch_b8)
    if (m0 + 32 <= M - 16) __builtin_prefetch(src + (size_t)32 * Kd, 0, 1);
  };

  // prologue: stage first tile
  stage_tile(0, mBase);
  async_wait0();
  __syncthreads();

  for (int mt = 0; mt < MT_PER_BLOCK; ++mt) {
    const int m0  = mBase + mt * 16;
    const int buf = mt & 1;

    // kick off async staging of the next tile into the other buffer
    if (mt + 1 < MT_PER_BLOCK) stage_tile(buf ^ 1, m0 + 16);

    // ---- 16x16x256 via 8 WMMAs, A fragments from LDS ----
    v8f acc = {};
    const __bf16* aRow = &sA[buf][l16 * CE];
#pragma unroll
    for (int ks = 0; ks < 8; ++ks) {
      // A k order used here (contiguous, same as B): half 0 -> K 0..15; half 1 -> K 16..31
      BF a;
      const int ka = ks * 32 + half * 16;
      a.h[0] = *(const v8bf*)(aRow + ka);
      a.h[1] = *(const v8bf*)(aRow + ka + 8);
      acc = __builtin_amdgcn_wmma_f32_16x16x32_bf16(
          /*neg_a=*/false, a.v, /*neg_b=*/false, bfrag[ks].v,
          /*c_mod=*/(short)0, acc, /*reuse_a=*/false, /*reuse_b=*/false);
    }

    // C/D layout: VGPR i, lanes 0-15 -> M=m0+i; lanes 16-31 -> M=m0+8+i.
    // Hoist the residual branch (kernel-uniform) and batch the 8 loads so
    // they issue as one clause under a single wait.
    const size_t cBase = (size_t)(m0 + half * 8) * N + nCol;
    if (resid) {
      float r[8];
#pragma unroll
      for (int i = 0; i < 8; ++i) r[i] = resid[cBase + (size_t)i * N];
#pragma unroll
      for (int i = 0; i < 8; ++i) C[cBase + (size_t)i * N] = acc[i] + r[i];
    } else {
#pragma unroll
      for (int i = 0; i < 8; ++i) C[cBase + (size_t)i * N] = acc[i];
    }

    async_wait0();     // our staging of the next buffer has landed in LDS
    __syncthreads();   // everyone done reading `buf`, next buffer visible
  }
}

// ---------------------------------------------------------------------------
// Kernel 4: gather neighbor features + direction vectors (bf16)
// One wave per (b,v,k) row; 32 lanes x 8 elems = E=256. 8 rows per block.
// ---------------------------------------------------------------------------
__global__ __launch_bounds__(256) void gather_kv_kernel(const __bf16* __restrict__ xbf,
                                                        const int* __restrict__ idx,
                                                        __bf16* __restrict__ kfeat,
                                                        __bf16* __restrict__ vin) {
  const int row  = blockIdx.x * 8 + (threadIdx.x >> 5);   // over B*V*K
  const int lane = threadIdx.x & 31;
  const int v = (row / CK) % CV;
  const int b =  row / (CK * CV);
  const int nb = idx[row];

  const __bf16* src  = xbf + ((size_t)b * CV + nb) * CE;
  const __bf16* self = xbf + ((size_t)b * CV + v) * CE;
  const int e = lane * 8;

  v8bf kv = *(const v8bf*)(src + e);
  v8bf sv = *(const v8bf*)(self + e);
  *(v8bf*)(kfeat + (size_t)row * CE + e) = kv;
  v8bf d;
#pragma unroll
  for (int i = 0; i < 8; ++i) d[i] = f2bf(bf2f(kv[i]) - bf2f(sv[i]));
  *(v8bf*)(vin + (size_t)row * CE + e) = d;
}

// ---------------------------------------------------------------------------
// Kernel 5: per-point attention (scores, softmax over K=8, context, head-mean)
// One thread per (b,v).
// ---------------------------------------------------------------------------
__global__ __launch_bounds__(256) void attn_kernel(const float* __restrict__ qh,
                                                   const float* __restrict__ kh,
                                                   const float* __restrict__ vh,
                                                   __bf16* __restrict__ ctx,
                                                   float* __restrict__ xwpk) {
  const int t = blockIdx.x * 256 + threadIdx.x;            // b*V + v
  const float* q    = qh + (size_t)t * CE;
  const float* krow = kh + (size_t)t * CK * CE;            // [K, E]
  const float* vrow = vh + (size_t)t * CK * CE;
  const float scale = 0.17677669529663687f;                // 1/sqrt(32)

  float wk[CK];
#pragma unroll
  for (int k = 0; k < CK; ++k) wk[k] = 0.f;

  for (int h = 0; h < CH; ++h) {
    float s[CK];
    float mx = -3.4e38f;
#pragma unroll
    for (int k = 0; k < CK; ++k) {
      float acc = 0.f;
      for (int d = 0; d < CHD; ++d)
        acc += q[h * CHD + d] * krow[k * CE + h * CHD + d];
      s[k] = acc * scale;
      mx = fmaxf(mx, s[k]);
    }
    float sum = 0.f;
#pragma unroll
    for (int k = 0; k < CK; ++k) { s[k] = __expf(s[k] - mx); sum += s[k]; }
    const float inv = 1.f / sum;
#pragma unroll
    for (int k = 0; k < CK; ++k) { s[k] *= inv; wk[k] += s[k] * (1.f / CH); }

    for (int d = 0; d < CHD; ++d) {
      float o = 0.f;
#pragma unroll
      for (int k = 0; k < CK; ++k) o += s[k] * vrow[k * CE + h * CHD + d];
      ctx[(size_t)t * CE + h * CHD + d] = f2bf(o);
    }
  }
#pragma unroll
  for (int k = 0; k < CK; ++k) xwpk[(size_t)t * CK + k] = wk[k];
}

// ---------------------------------------------------------------------------
// Kernels 6-8: scatter-mean
// ---------------------------------------------------------------------------
__global__ __launch_bounds__(256) void zero_kernel(float* __restrict__ p, int n) {
  int i = blockIdx.x * 256 + threadIdx.x;
  if (i < n) p[i] = 0.f;
}

__global__ __launch_bounds__(256) void scatter_kernel(const int* __restrict__ idx,
                                                      const float* __restrict__ xwpk,
                                                      float* __restrict__ sums,
                                                      float* __restrict__ counts) {
  const int t = blockIdx.x * 256 + threadIdx.x;            // over B*V*K
  const int b = t / (CV * CK);
  const int tgt = idx[t];
  atomicAdd(&sums[b * CV + tgt], xwpk[t]);
  atomicAdd(&counts[b * CV + tgt], 1.0f);
}

__global__ __launch_bounds__(256) void finalize_xw_kernel(const float* __restrict__ sums,
                                                          const float* __restrict__ counts,
                                                          float* __restrict__ out) {
  const int t = blockIdx.x * 256 + threadIdx.x;            // over B*V
  out[t] = sums[t] / (counts[t] + 1.0f);
}

// ---------------------------------------------------------------------------
// Launcher
// ---------------------------------------------------------------------------
extern "C" void kernel_launch(void* const* d_in, const int* in_sizes, int n_in,
                              void* d_out, int out_size, void* d_ws, size_t ws_size,
                              hipStream_t stream) {
  (void)in_sizes; (void)n_in; (void)out_size; (void)ws_size;
  const float* x  = (const float*)d_in[0];
  const float* xv = (const float*)d_in[1];
  const float* Wq = (const float*)d_in[2];
  const float* Wk = (const float*)d_in[3];
  const float* Wv = (const float*)d_in[4];
  const float* Wo = (const float*)d_in[5];

  char* ws = (char*)d_ws;
  size_t off = 0;
  auto take = [&](size_t bytes) -> void* {
    void* p = ws + off;
    off += (bytes + 255) & ~(size_t)255;
    return p;
  };

  const size_t BV  = (size_t)CB * CV;          // 16384
  const size_t BVK = BV * CK;                  // 131072

  int*    idx    = (int*)   take(BVK * sizeof(int));
  __bf16* xbf    = (__bf16*)take(BV * CE * sizeof(__bf16));
  __bf16* wqb    = (__bf16*)take((size_t)CE * CE * sizeof(__bf16));
  __bf16* wkb    = (__bf16*)take((size_t)CE * CE * sizeof(__bf16));
  __bf16* wvb    = (__bf16*)take((size_t)CE * CE * sizeof(__bf16));
  __bf16* wob    = (__bf16*)take((size_t)CE * CE * sizeof(__bf16));
  float*  qh     = (float*) take(BV * CE * sizeof(float));
  __bf16* kfeat  = (__bf16*)take(BVK * CE * sizeof(__bf16));
  __bf16* vin    = (__bf16*)take(BVK * CE * sizeof(__bf16));
  float*  kh     = (float*) take(BVK * CE * sizeof(float));
  float*  vh     = (float*) take(BVK * CE * sizeof(float));
  __bf16* ctx    = (__bf16*)take(BV * CE * sizeof(__bf16));
  float*  xwpk   = (float*) take(BVK * sizeof(float));
  float*  sums   = (float*) take(BV * sizeof(float));     // contiguous with counts
  float*  counts = (float*) take(BV * sizeof(float));

  float* out_main = (float*)d_out;             // x + out : [B,V,E]
  float* out_xw   = out_main + BV * CE;        // x_w     : [B,V,1]

  // 1) KNN
  knn_kernel<<<CB * (CV / 256), 256, 0, stream>>>(xv, idx);

  // 2) fp32 -> bf16
  {
    const int nx = (int)(BV * CE);
    cvt_bf16_kernel<<<(nx + 255) / 256, 256, 0, stream>>>(x,  xbf, nx);
    const int nw = CE * CE;
    cvt_bf16_kernel<<<(nw + 255) / 256, 256, 0, stream>>>(Wq, wqb, nw);
    cvt_bf16_kernel<<<(nw + 255) / 256, 256, 0, stream>>>(Wk, wkb, nw);
    cvt_bf16_kernel<<<(nw + 255) / 256, 256, 0, stream>>>(Wv, wvb, nw);
    cvt_bf16_kernel<<<(nw + 255) / 256, 256, 0, stream>>>(Wo, wob, nw);
  }

  // 3) qh = x @ Wq^T           M=16384, N=256, K=256
  wmma_gemm_bf16<<<dim3(CE / 128, (unsigned)(BV / (16 * MT_PER_BLOCK))), 256, 0, stream>>>(
      xbf, wqb, qh, nullptr, (int)BV, CE, CE);

  // 4) gather kfeat / v_in
  gather_kv_kernel<<<(unsigned)(BVK / 8), 256, 0, stream>>>(xbf, idx, kfeat, vin);

  // 5) kh = kfeat @ Wk^T ; vh = v_in @ Wv^T   M=131072
  wmma_gemm_bf16<<<dim3(CE / 128, (unsigned)(BVK / (16 * MT_PER_BLOCK))), 256, 0, stream>>>(
      kfeat, wkb, kh, nullptr, (int)BVK, CE, CE);
  wmma_gemm_bf16<<<dim3(CE / 128, (unsigned)(BVK / (16 * MT_PER_BLOCK))), 256, 0, stream>>>(
      vin, wvb, vh, nullptr, (int)BVK, CE, CE);

  // 6) attention
  attn_kernel<<<(unsigned)(BV / 256), 256, 0, stream>>>(qh, kh, vh, ctx, xwpk);

  // 7) out = ctx @ Wo^T + x  -> d_out[0 : B*V*E]
  wmma_gemm_bf16<<<dim3(CE / 128, (unsigned)(BV / (16 * MT_PER_BLOCK))), 256, 0, stream>>>(
      ctx, wob, out_main, x, (int)BV, CE, CE);

  // 8) scatter-mean -> d_out[B*V*E : +B*V]
  zero_kernel<<<(unsigned)((2 * BV + 255) / 256), 256, 0, stream>>>(sums, (int)(2 * BV));
  scatter_kernel<<<(unsigned)(BVK / 256), 256, 0, stream>>>(idx, xwpk, sums, counts);
  finalize_xw_kernel<<<(unsigned)(BV / 256), 256, 0, stream>>>(sums, counts, out_xw);
}
